// GPT_22857815949576
// MI455X (gfx1250) — matrix operand order repaired and checked by reading, not compiled, not run on hardware
//
#include <hip/hip_runtime.h>
#include <hip/hip_bf16.h>
#include <math.h>

// ---------------- problem constants (GPT-2 small fwd) ----------------
#define BB 2
#define TT 1024
#define VV 50257
#define CC 768
#define LL 12
#define HH 12
#define HD 64
#define NT (BB * TT)          // 2048 token rows

// ---------------- WMMA vector types (gfx1250, wave32) ----------------
typedef __attribute__((ext_vector_type(16))) __bf16 v16bf;
typedef __attribute__((ext_vector_type(8)))  float  v8f;
typedef __attribute__((ext_vector_type(4)))  unsigned int u32x4;
typedef __attribute__((ext_vector_type(4)))  int          i32x4;
typedef __attribute__((ext_vector_type(8)))  int          i32x8;

union AF { v16bf v; unsigned u[8]; };

__device__ __forceinline__ unsigned short f2bf(float f) {
    union { float f; unsigned u; } c{f};
    unsigned r = c.u + 0x7fffu + ((c.u >> 16) & 1u);   // round-to-nearest-even
    return (unsigned short)(r >> 16);
}

__device__ __forceinline__ float gelu_tanh(float x) {
    float x3 = x * x * x;
    return 0.5f * x * (1.f + tanhf(0.7978845608028654f * (x + 0.044715f * x3)));
}

// ---------------------------------------------------------------------
// Tensor Data Mover: 2-D tile (rows x 32 bf16 cols) global -> LDS.
// D# built per CDNA5 ISA 8.3/8.4: group0 {count, lds_addr, global_addr,
// type=2}, group1 {data_size=2B, tensor dims, tile dims, dim0 stride}.
// Arity differs between toolchains (probe-verified); the TDM header is
// the documented discriminator.
// ---------------------------------------------------------------------
__device__ __forceinline__ void tdm_load_tile_bf16(
    unsigned lds_off, const unsigned short* gptr,
    unsigned rows, unsigned rowStrideElems, unsigned tensorDim0)
{
    unsigned long long ga = (unsigned long long)(size_t)gptr;
    u32x4 g0;
    g0.x = 1u;                                   // count=1, user descriptor
    g0.y = lds_off;                              // LDS byte address
    g0.z = (unsigned)ga;                         // global_addr[31:0]
    g0.w = (unsigned)((ga >> 32) & 0x01FFFFFFull) | (2u << 30);  // [56:32] | type=2

    unsigned td0 = tensorDim0;                   // tensor_dim0 (elements)
    unsigned td1 = 1u << 20;                     // tensor_dim1 (large)
    i32x8 g1;
    g1[0] = (int)(1u << 16);                     // workgroup_mask=0, data_size=1 (2B)
    g1[1] = (int)((td0 & 0xffffu) << 16);        // atomic_addr=0 | tensor_dim0[15:0]
    g1[2] = (int)((td0 >> 16) | ((td1 & 0xffffu) << 16));  // dim0 hi | dim1 lo
    g1[3] = (int)((td1 >> 16) | (32u << 16));    // dim1 hi | tile_dim0=32
    g1[4] = (int)rows;                           // tile_dim1=rows, tile_dim2=0
    g1[5] = (int)rowStrideElems;                 // tensor_dim0_stride[31:0]
    g1[6] = 0;                                   // stride0 hi | stride1 lo
    g1[7] = 0;                                   // stride1 hi

    i32x4 g2; g2[0] = 1; g2[1] = 1; g2[2] = 0; g2[3] = 0;  // tensor_dim2/3=1
    i32x4 g3; g3[0] = 0; g3[1] = 0; g3[2] = 0; g3[3] = 0;

#if __has_include(<hip/amd_detail/amd_gfx1250_TDM.h>)
    i32x8 g4; g4[0]=0; g4[1]=0; g4[2]=0; g4[3]=0; g4[4]=0; g4[5]=0; g4[6]=0; g4[7]=0;
    __builtin_amdgcn_tensor_load_to_lds(g0, g1, g2, g3, g4, 0);
#else
    __builtin_amdgcn_tensor_load_to_lds(g0, g1, g2, g3, 0);
#endif
}

// ---------------- GEMM epilogue modes ----------------
#define EPI_STORE     0   // out f32 = acc
#define EPI_BIAS      1   // out f32 = acc + bias
#define EPI_BIAS_RES  2   // out f32 = acc + bias + res
#define EPI_BIAS_GELU 3   // out bf16 = gelu(acc + bias)
#define EPI_SCALE     4   // out f32 = acc * scale

// =====================================================================
// Tiled WMMA GEMM:  out[M,N] = A_bf16[M,K] x B[K,N]  (+ epilogue)
//   Block tile 128(M) x 64(N), 8 waves, each wave 32x32 (2x2 WMMA tiles),
//   K stepped by 32 through LDS.  A tile staged by the TDM (wave 0);
//   bf16 [N,K] B tiles staged with per-lane async global->LDS copies.
//   BTRANS: B stored [N,K]. BBF16: B is bf16. Batched over blockIdx.z.
// =====================================================================
template<int BTRANS, int BBF16, int EPI>
__global__ void __launch_bounds__(256)
gemm_bf16_wmma(const unsigned short* __restrict__ A,
               const void*           __restrict__ Bw,
               const float*          __restrict__ bias,
               const float*          __restrict__ res,
               float*                __restrict__ outF,
               unsigned short*       __restrict__ outB,
               int M, int N, int K, float scale,
               long long aZ, long long bZ, long long oZ)
{
    __shared__ __align__(16) unsigned short lds_a[128 * 32];  // A tile, row-major [m][k]
    __shared__ __align__(16) unsigned short lds_b[64 * 32];   // B tile, n-major  [n][k]

    const int tid = threadIdx.x;
    const int z   = blockIdx.z;
    A += (size_t)z * aZ;
    const float*          Bf = (const float*)Bw          + (size_t)z * bZ;
    const unsigned short* Bh = (const unsigned short*)Bw + (size_t)z * bZ;
    outF += (size_t)z * oZ;
    if (EPI == EPI_BIAS_GELU) outB += (size_t)z * oZ;

    const int bM = blockIdx.y * 128;
    const int bN = blockIdx.x * 64;

    const int wave = tid >> 5, lane = tid & 31;
    const int wm = wave >> 1, wn = wave & 1;        // 4x2 wave grid
    const int hl = lane >> 4, lm = lane & 15;       // half / in-half lane

    const unsigned ldsA_base = (unsigned)(size_t)(void*)lds_a;  // LDS aperture low bits

    v8f acc[2][2];
    #pragma unroll
    for (int i = 0; i < 2; i++)
        #pragma unroll
        for (int j = 0; j < 2; j++)
            #pragma unroll
            for (int e = 0; e < 8; e++) acc[i][j][e] = 0.f;

    for (int k0 = 0; k0 < K; k0 += 32) {
        if (k0 + 32 < K)  // gfx1250 global_prefetch_b8 of next A slab
            __builtin_prefetch(&A[(size_t)(bM + (tid >> 1)) * K + k0 + 32], 0, 1);

        // ---- stage A tile 128x32 bf16 via Tensor Data Mover (wave 0) ----
        if (wave == 0)
            tdm_load_tile_bf16(ldsA_base, &A[(size_t)bM * K + k0],
                               128u, (unsigned)K, (unsigned)K);

        // ---- stage B tile -> lds_b[n][k] ----
        if (BTRANS == 0 && BBF16 == 0) {
            // f32 [K,N] row-major weights (N multiple of 64): load+convert
            #pragma unroll
            for (int it = 0; it < 2; it++) {
                int e4 = tid + it * 256;
                int kk = e4 >> 4, nn4 = e4 & 15;
                const float4 f = *(const float4*)&Bf[(size_t)(k0 + kk) * N + bN + nn4 * 4];
                lds_b[(nn4 * 4 + 0) * 32 + kk] = f2bf(f.x);
                lds_b[(nn4 * 4 + 1) * 32 + kk] = f2bf(f.y);
                lds_b[(nn4 * 4 + 2) * 32 + kk] = f2bf(f.z);
                lds_b[(nn4 * 4 + 3) * 32 + kk] = f2bf(f.w);
            }
        } else if (BTRANS == 1 && BBF16 == 0) {
            // f32 [N,K] (torch Linear layout, e.g. lm_head [V,C]); N-bounds checked
            #pragma unroll
            for (int it = 0; it < 2; it++) {
                int e4 = tid + it * 256;
                int n = e4 >> 3, c4 = e4 & 7;
                int gn = bN + n;
                float4 f; f.x = f.y = f.z = f.w = 0.f;
                if (gn < N) f = *(const float4*)&Bf[(size_t)gn * K + k0 + c4 * 4];
                lds_b[n * 32 + c4 * 4 + 0] = f2bf(f.x);
                lds_b[n * 32 + c4 * 4 + 1] = f2bf(f.y);
                lds_b[n * 32 + c4 * 4 + 2] = f2bf(f.z);
                lds_b[n * 32 + c4 * 4 + 3] = f2bf(f.w);
            }
        } else if (BTRANS == 1 && BBF16 == 1) {
            // bf16 [N,K] (K matrix for Q*K^T): per-lane async global->LDS b128
            {
                int n = tid >> 2, c8 = tid & 3;                   // 64 rows x 4 chunks
                unsigned ldsoff = (unsigned)(size_t)(void*)&lds_b[n * 32 + c8 * 8];
                const unsigned short* gp = &Bh[(size_t)(bN + n) * K + k0 + c8 * 8];
                asm volatile("global_load_async_to_lds_b128 %0, %1, off"
                             :: "v"(ldsoff), "v"(gp) : "memory");
            }
        } else {
            // bf16 [K,N] (V matrix for P*V, N=64): manual transpose store
            #pragma unroll
            for (int it = 0; it < 4; it++) {
                int e2 = tid + it * 256;
                int kk = e2 >> 5, nn2 = e2 & 31;
                unsigned u = *(const unsigned*)&Bh[(size_t)(k0 + kk) * N + bN + nn2 * 2];
                lds_b[(nn2 * 2 + 0) * 32 + kk] = (unsigned short)(u & 0xffffu);
                lds_b[(nn2 * 2 + 1) * 32 + kk] = (unsigned short)(u >> 16);
            }
        }

        // ---- drain data movers, then rendezvous ----
        if (BTRANS == 1 && BBF16 == 1)
            asm volatile("s_wait_asynccnt 0x0" ::: "memory");
        if (wave == 0)
            __builtin_amdgcn_s_wait_tensorcnt((short)0);
        __syncthreads();

        // ---- fragments per ISA layout tables ----
        AF a[2], b2[2];
        #pragma unroll
        for (int i = 0; i < 2; i++)
            #pragma unroll
            for (int v = 0; v < 8; v++) {
                // 16-bit A 16x32: K = (v>=4?16:0) + half*8 + (v&3)*2
                int ka = ((v & 4) << 2) + (hl << 3) + ((v & 3) << 1);
                a[i].u[v] = *(const unsigned*)&lds_a[(wm * 32 + i * 16 + lm) * 32 + ka];
            }
        #pragma unroll
        for (int j = 0; j < 2; j++)
            #pragma unroll
            for (int v = 0; v < 8; v++) {
                // 16-bit B 32x16: K = (v>=4?16:0) + half*8 + (v&3)*2
                int kb = ((v & 4) << 2) + (hl << 3) + ((v & 3) << 1);
                b2[j].u[v] = *(const unsigned*)&lds_b[(wn * 32 + j * 16 + lm) * 32 + kb];
            }
        #pragma unroll
        for (int i = 0; i < 2; i++)
            #pragma unroll
            for (int j = 0; j < 2; j++)
                acc[i][j] = __builtin_amdgcn_wmma_f32_16x16x32_bf16(
                    false, a[i].v, false, b2[j].v, (short)0, acc[i][j], false, false);
        __syncthreads();
    }

    // ---- epilogue: C/D layout => row = half*8 + vgpr, col = lane&15 ----
    #pragma unroll
    for (int j = 0; j < 2; j++) {
        int gn = bN + wn * 32 + j * 16 + lm;
        if (gn >= N) continue;
        float bv = (EPI == EPI_BIAS || EPI == EPI_BIAS_RES || EPI == EPI_BIAS_GELU)
                   ? bias[gn] : 0.f;
        #pragma unroll
        for (int i = 0; i < 2; i++) {
            int gm0 = bM + wm * 32 + i * 16 + hl * 8;
            #pragma unroll
            for (int e = 0; e < 8; e++) {
                int gm = gm0 + e;
                float v = acc[i][j][e];
                if (EPI == EPI_SCALE) v *= scale;
                v += bv;
                if (EPI == EPI_BIAS_RES) v += res[(size_t)gm * N + gn];
                if (EPI == EPI_BIAS_GELU)
                    outB[(size_t)gm * N + gn] = f2bf(gelu_tanh(v));
                else
                    outF[(size_t)gm * N + gn] = v;
            }
        }
    }
}

// =====================================================================
// Elementwise / reduction glue kernels
// =====================================================================
__global__ void __launch_bounds__(256)
embed_kernel(const int* __restrict__ idx, const float* __restrict__ wte,
             const float* __restrict__ wpe, float* __restrict__ x)
{
    int i = blockIdx.x * 256 + threadIdx.x;        // NT*CC total (divisible)
    int bt = i / CC, c = i - bt * CC;
    int t = bt & (TT - 1);
    x[i] = wte[(size_t)idx[bt] * CC + c] + wpe[(size_t)t * CC + c];
}

__global__ void __launch_bounds__(256)
layernorm_bf16_kernel(const float* __restrict__ x, const float* __restrict__ w,
                      const float* __restrict__ b, unsigned short* __restrict__ out)
{
    __shared__ float red[8];
    const int row = blockIdx.x, tid = threadIdx.x, lane = tid & 31, wid = tid >> 5;
    const float* xr = x + (size_t)row * CC;
    float v0 = xr[tid], v1 = xr[tid + 256], v2 = xr[tid + 512];
    float s = v0 + v1 + v2;
    #pragma unroll
    for (int o = 16; o; o >>= 1) s += __shfl_xor(s, o, 32);
    if (!lane) red[wid] = s;
    __syncthreads();
    if (!tid) { float t = 0.f; for (int i = 0; i < 8; i++) t += red[i]; red[0] = t * (1.f / CC); }
    __syncthreads();
    float mean = red[0];
    __syncthreads();
    float d0 = v0 - mean, d1 = v1 - mean, d2 = v2 - mean;
    float s2 = d0 * d0 + d1 * d1 + d2 * d2;
    #pragma unroll
    for (int o = 16; o; o >>= 1) s2 += __shfl_xor(s2, o, 32);
    if (!lane) red[wid] = s2;
    __syncthreads();
    if (!tid) { float t = 0.f; for (int i = 0; i < 8; i++) t += red[i]; red[0] = t * (1.f / CC); }
    __syncthreads();
    float rinv = rsqrtf(red[0] + 1e-5f);
    size_t o0 = (size_t)row * CC;
    out[o0 + tid]       = f2bf(d0 * rinv * w[tid]       + b[tid]);
    out[o0 + tid + 256] = f2bf(d1 * rinv * w[tid + 256] + b[tid + 256]);
    out[o0 + tid + 512] = f2bf(d2 * rinv * w[tid + 512] + b[tid + 512]);
}

__global__ void __launch_bounds__(256)
split_qkv_kernel(const float* __restrict__ qkv, unsigned short* __restrict__ qb,
                 unsigned short* __restrict__ kb, unsigned short* __restrict__ vb)
{
    int i = blockIdx.x * 256 + threadIdx.x;        // NT*3*CC total (divisible)
    int bt = i / (3 * CC), r = i - bt * (3 * CC);
    int s = r / CC, c = r - s * CC;
    int h = c >> 6, d = c & 63;
    int bq = bt / TT, t = bt - bq * TT;
    size_t dst = ((size_t)(bq * HH + h) * TT + t) * HD + d;   // [B,H,T,HD]
    unsigned short w = f2bf(qkv[i]);
    if (s == 0) qb[dst] = w; else if (s == 1) kb[dst] = w; else vb[dst] = w;
}

__global__ void __launch_bounds__(256)
softmax_causal_bf16(const float* __restrict__ att, unsigned short* __restrict__ attp)
{
    const int q = blockIdx.x;
    const size_t base = ((size_t)blockIdx.y * TT + q) * TT;
    const float* row = att + base;
    unsigned short* orow = attp + base;
    __shared__ float red[8];
    const int tid = threadIdx.x, lane = tid & 31, wid = tid >> 5;
    float v[4]; float mx = -3.4e38f;
    #pragma unroll
    for (int j = 0; j < 4; j++) {
        int c = tid + j * 256;
        float xv = (c <= q) ? row[c] : -3.4e38f;
        v[j] = xv; mx = fmaxf(mx, xv);
    }
    #pragma unroll
    for (int o = 16; o; o >>= 1) mx = fmaxf(mx, __shfl_xor(mx, o, 32));
    if (!lane) red[wid] = mx;
    __syncthreads();
    if (!tid) { float m = red[0]; for (int i = 1; i < 8; i++) m = fmaxf(m, red[i]); red[0] = m; }
    __syncthreads();
    mx = red[0];
    __syncthreads();
    float s = 0.f;
    #pragma unroll
    for (int j = 0; j < 4; j++) {
        int c = tid + j * 256;
        float e = (c <= q) ? __expf(v[j] - mx) : 0.f;
        v[j] = e; s += e;
    }
    #pragma unroll
    for (int o = 16; o; o >>= 1) s += __shfl_xor(s, o, 32);
    if (!lane) red[wid] = s;
    __syncthreads();
    if (!tid) { float t = 0.f; for (int i = 0; i < 8; i++) t += red[i]; red[0] = t; }
    __syncthreads();
    const float inv = 1.f / red[0];
    #pragma unroll
    for (int j = 0; j < 4; j++) {
        int c = tid + j * 256;
        orow[c] = f2bf(v[j] * inv);      // masked cols -> exact 0 (safe for P*V)
    }
}

__global__ void __launch_bounds__(256)
merge_y_kernel(const float* __restrict__ yh, unsigned short* __restrict__ ybf)
{
    int i = blockIdx.x * 256 + threadIdx.x;        // NT*CC total
    int bt = i / CC, c = i - bt * CC;
    int h = c >> 6, d = c & 63;
    int bq = bt / TT, t = bt - bq * TT;
    ybf[i] = f2bf(yh[((size_t)(bq * HH + h) * TT + t) * HD + d]);
}

// =====================================================================
// Orchestration
// =====================================================================
extern "C" void kernel_launch(void* const* d_in, const int* in_sizes, int n_in,
                              void* d_out, int out_size, void* d_ws, size_t ws_size,
                              hipStream_t stream)
{
    (void)in_sizes; (void)n_in; (void)out_size; (void)ws_size;
    const int*   idx   = (const int*)  d_in[0];
    const float* wte   = (const float*)d_in[1];
    const float* wpe   = (const float*)d_in[2];
    const float* ln1w  = (const float*)d_in[3];
    const float* ln1b  = (const float*)d_in[4];
    const float* attw  = (const float*)d_in[5];
    const float* attb  = (const float*)d_in[6];
    const float* projw = (const float*)d_in[7];
    const float* projb = (const float*)d_in[8];
    const float* ln2w  = (const float*)d_in[9];
    const float* ln2b  = (const float*)d_in[10];
    const float* fcw   = (const float*)d_in[11];
    const float* fcb   = (const float*)d_in[12];
    const float* fc2w  = (const float*)d_in[13];
    const float* fc2b  = (const float*)d_in[14];
    const float* lnfw  = (const float*)d_in[15];
    const float* lnfb  = (const float*)d_in[16];
    const float* lmw   = (const float*)d_in[17];
    float* logits = (float*)d_out;

    char* ws = (char*)d_ws;
    size_t off = 0;
    auto take = [&](size_t bytes) -> char* {
        char* p = ws + off;
        off = (off + bytes + 255) & ~(size_t)255;
        return p;
    };
    float*          x    = (float*)         take((size_t)NT * CC * 4);
    unsigned short* h    = (unsigned short*)take((size_t)NT * CC * 2);
    float*          qkv  = (float*)         take((size_t)NT * 3 * CC * 4);
    unsigned short* qb   = (unsigned short*)take((size_t)NT * CC * 2);
    unsigned short* kb   = (unsigned short*)take((size_t)NT * CC * 2);
    unsigned short* vb   = (unsigned short*)take((size_t)NT * CC * 2);
    float*          att  = (float*)         take((size_t)BB * HH * TT * TT * 4);
    unsigned short* attp = (unsigned short*)take((size_t)BB * HH * TT * TT * 2);
    float*          yh   = (float*)         take((size_t)NT * CC * 4);
    unsigned short* ybf  = (unsigned short*)take((size_t)NT * CC * 2);
    unsigned short* gel  = (unsigned short*)take((size_t)NT * 4 * CC * 2);

    const dim3 blk(256);

    embed_kernel<<<NT * CC / 256, blk, 0, stream>>>(idx, wte, wpe, x);

    for (int l = 0; l < LL; l++) {
        layernorm_bf16_kernel<<<NT, blk, 0, stream>>>(x, ln1w + l * CC, ln1b + l * CC, h);

        // QKV: [2048,768] x [768,2304] + bias
        gemm_bf16_wmma<0, 0, EPI_BIAS><<<dim3(3 * CC / 64, NT / 128, 1), blk, 0, stream>>>(
            h, attw + (size_t)l * CC * 3 * CC, attb + (size_t)l * 3 * CC,
            nullptr, qkv, nullptr, NT, 3 * CC, CC, 1.f, 0, 0, 0);

        split_qkv_kernel<<<NT * 3 * CC / 256, blk, 0, stream>>>(qkv, qb, kb, vb);

        // scores = Q x K^T * (1/sqrt(64)), batched over B*H
        gemm_bf16_wmma<1, 1, EPI_SCALE><<<dim3(TT / 64, TT / 128, BB * HH), blk, 0, stream>>>(
            qb, kb, nullptr, nullptr, att, nullptr,
            TT, TT, HD, 0.125f,
            (long long)TT * HD, (long long)TT * HD, (long long)TT * TT);

        softmax_causal_bf16<<<dim3(TT, BB * HH), blk, 0, stream>>>(att, attp);

        // y = P x V, batched over B*H
        gemm_bf16_wmma<0, 1, EPI_STORE><<<dim3(1, TT / 128, BB * HH), blk, 0, stream>>>(
            attp, vb, nullptr, nullptr, yh, nullptr,
            TT, HD, TT, 1.f,
            (long long)TT * TT, (long long)TT * HD, (long long)TT * HD);

        merge_y_kernel<<<NT * CC / 256, blk, 0, stream>>>(yh, ybf);

        // x = x + y @ proj_w + proj_b
        gemm_bf16_wmma<0, 0, EPI_BIAS_RES><<<dim3(CC / 64, NT / 128, 1), blk, 0, stream>>>(
            ybf, projw + (size_t)l * CC * CC, projb + (size_t)l * CC,
            x, x, nullptr, NT, CC, CC, 1.f, 0, 0, 0);

        layernorm_bf16_kernel<<<NT, blk, 0, stream>>>(x, ln2w + l * CC, ln2b + l * CC, h);

        // fc + GELU (bf16 out)
        gemm_bf16_wmma<0, 0, EPI_BIAS_GELU><<<dim3(4 * CC / 64, NT / 128, 1), blk, 0, stream>>>(
            h, fcw + (size_t)l * CC * 4 * CC, fcb + (size_t)l * 4 * CC,
            nullptr, nullptr, gel, NT, 4 * CC, CC, 1.f, 0, 0, 0);

        // x = x + gelu @ fc2_w + fc2_b
        gemm_bf16_wmma<0, 0, EPI_BIAS_RES><<<dim3(CC / 64, NT / 128, 1), blk, 0, stream>>>(
            gel, fc2w + (size_t)l * 4 * CC * CC, fc2b + (size_t)l * CC,
            x, x, nullptr, NT, CC, 4 * CC, 1.f, 0, 0, 0);
    }

    layernorm_bf16_kernel<<<NT, blk, 0, stream>>>(x, lnfw, lnfb, h);

    // logits = lnf(x) @ lm_head_w^T   ([2048,768] x [50257,768]^T)
    gemm_bf16_wmma<1, 0, EPI_STORE><<<dim3((VV + 63) / 64, NT / 128, 1), blk, 0, stream>>>(
        h, lmw, nullptr, nullptr, logits, nullptr,
        NT, VV, CC, 1.f, 0, 0, 0);
}
